// _StreamingLLMAttention_44890998177946
// MI455X (gfx1250) — matrix operand order repaired and checked by reading, not compiled, not run on hardware
//
#include <hip/hip_runtime.h>
#include <hip/hip_bf16.h>

// ---------------- problem constants ----------------
#define SEQ     4096
#define DMODEL  1024
#define NHEADS  16
#define HDIM    64
#define WINDOW  512
#define NSINK   4

typedef __attribute__((ext_vector_type(16))) __bf16          v16bf;
typedef __attribute__((ext_vector_type(8)))  __bf16          v8bf;
typedef __attribute__((ext_vector_type(8)))  float           v8f;
typedef __attribute__((ext_vector_type(4)))  float           f4;
typedef __attribute__((ext_vector_type(8)))  unsigned short  v8us;

// ---------------- bf16 split helpers (round-to-nearest-even) ----------------
static __device__ __forceinline__ unsigned short f2bf_rne(float f) {
  unsigned u = __float_as_uint(f);
  return (unsigned short)((u + 0x7FFFu + ((u >> 16) & 1u)) >> 16);
}
static __device__ __forceinline__ float bfbits2f(unsigned short b) {
  return __uint_as_float(((unsigned)b) << 16);
}
static __device__ __forceinline__ __bf16 bfb(unsigned short b) {
  return __builtin_bit_cast(__bf16, b);
}

// 16 consecutive bf16 (32B aligned) -> full WMMA operand
static __device__ __forceinline__ v16bf ld16(const unsigned short* p) {
  return *(const v16bf*)p;
}
// two 8-element chunks (16B aligned each) -> WMMA operand
static __device__ __forceinline__ v16bf ld8x2(const unsigned short* p0, const unsigned short* p1) {
  v8bf a = *(const v8bf*)p0;
  v8bf b = *(const v8bf*)p1;
  return __builtin_shufflevector(a, b, 0, 1, 2, 3, 4, 5, 6, 7, 8, 9, 10, 11, 12, 13, 14, 15);
}

#define WMMA_BF16(A, B, C) \
  __builtin_amdgcn_wmma_f32_16x16x32_bf16(false, (A), false, (B), (short)0, (C), false, false)

// ============================================================================
// One-time elementwise split: fp32 -> bf16 hi/lo planes. 8 elems / thread.
// ============================================================================
__global__ void __launch_bounds__(256)
split_f32_bf16x2(const float* __restrict__ in, unsigned short* __restrict__ hp,
                 unsigned short* __restrict__ lp, int n8) {
  int idx = blockIdx.x * blockDim.x + threadIdx.x;
  if (idx >= n8) return;
  f4 a = *((const f4*)in + idx * 2);
  f4 b = *((const f4*)in + idx * 2 + 1);
  float v[8] = {a[0], a[1], a[2], a[3], b[0], b[1], b[2], b[3]};
  v8us h, l;
#pragma unroll
  for (int i = 0; i < 8; ++i) {
    unsigned short hb = f2bf_rne(v[i]);
    h[i] = hb;
    l[i] = f2bf_rne(v[i] - bfbits2f(hb));
  }
  *(v8us*)(hp + (size_t)idx * 8) = h;
  *(v8us*)(lp + (size_t)idx * 8) = l;
}

// ============================================================================
// GEMM (NT) on pre-split planes: out = A(4096x1024) @ W(1024x1024)^T, bf16x3.
// One wave computes a 32(M) x 64(N) tile. WMMAs issued term-major so every
// accumulator is re-used at distance 8 (>= 5 slot bf16 WMMA RAW hazard).
// mode 0: fp32 out[m*1024+n]
// mode 1: bf16 hi/lo planes, out[m*1024+n]
// mode 2: bf16 hi/lo planes transposed, out[n*4096+m]   (V^T)
// ============================================================================
__global__ void __launch_bounds__(256)
gemm_nt_bf16x3(const unsigned short* __restrict__ Ah, const unsigned short* __restrict__ Al,
               const unsigned short* __restrict__ Bh, const unsigned short* __restrict__ Bl,
               float* __restrict__ outF, unsigned short* __restrict__ outH,
               unsigned short* __restrict__ outL, int mode) {
  int wave = blockIdx.x * (blockDim.x >> 5) + (threadIdx.x >> 5);
  int lane = threadIdx.x & 31;
  int li = lane & 15;
  bool hiH = lane >= 16;

  int mt = wave & 127;   // 128 M tiles of 32
  int nt = wave >> 7;    // 16 N groups of 64
  int m0 = mt * 32, n0 = nt * 64;

  v8f acc[2][4] = {};

  for (int k0 = 0; k0 < DMODEL; k0 += 32) {
    // A operands: rows m0+s*16+li; K = k0 + (lanes<16?0:16) + 0..15
    v16bf ah[2], al[2];
    int c0 = k0 + (hiH ? 16 : 0);
#pragma unroll
    for (int s = 0; s < 2; ++s) {
      const unsigned short* rh = Ah + (size_t)(m0 + s * 16 + li) * DMODEL + c0;
      const unsigned short* rl = Al + (size_t)(m0 + s * 16 + li) * DMODEL + c0;
      ah[s] = ld16(rh);
      al[s] = ld16(rl);
    }
    // B operands: col n holds W row n, K = k0 + (lanes<16?0:16) + 0..15
    v16bf bh[4], bl[4];
#pragma unroll
    for (int t = 0; t < 4; ++t) {
      size_t boff = (size_t)(n0 + t * 16 + li) * DMODEL + k0 + (hiH ? 16 : 0);
      bh[t] = ld16(Bh + boff);
      bl[t] = ld16(Bl + boff);
    }
    // term-major issue: 8 independent WMMAs between accumulator re-uses
#pragma unroll
    for (int t = 0; t < 4; ++t)
#pragma unroll
      for (int s = 0; s < 2; ++s) acc[s][t] = WMMA_BF16(ah[s], bh[t], acc[s][t]);
#pragma unroll
    for (int t = 0; t < 4; ++t)
#pragma unroll
      for (int s = 0; s < 2; ++s) acc[s][t] = WMMA_BF16(ah[s], bl[t], acc[s][t]);
#pragma unroll
    for (int t = 0; t < 4; ++t)
#pragma unroll
      for (int s = 0; s < 2; ++s) acc[s][t] = WMMA_BF16(al[s], bh[t], acc[s][t]);
  }

#pragma unroll
  for (int s = 0; s < 2; ++s) {
#pragma unroll
    for (int t = 0; t < 4; ++t) {
      int n = n0 + t * 16 + li;
#pragma unroll
      for (int r = 0; r < 8; ++r) {
        int m = m0 + s * 16 + (hiH ? 8 : 0) + r;  // C layout: vgpr r -> M=r (+8 hi half)
        float v = acc[s][t][r];
        if (mode == 0) {
          outF[(size_t)m * DMODEL + n] = v;
        } else {
          unsigned short hb = f2bf_rne(v);
          unsigned short lb = f2bf_rne(v - bfbits2f(hb));
          size_t o = (mode == 1) ? ((size_t)m * DMODEL + n) : ((size_t)n * SEQ + m);
          outH[o] = hb;
          outL[o] = lb;
        }
      }
    }
  }
}

// ============================================================================
// Streaming (sink + sliding window) flash attention on pre-split planes.
// Grid (SEQ/16, NHEADS), 1 wave. S^T = K x Q^T with 6 independent term
// accumulators (hazard-free round-robin); O^T = V^T x P^T term-major.
// ============================================================================
__global__ void __launch_bounds__(32)
attn_streaming_wmma(const unsigned short* __restrict__ qhp, const unsigned short* __restrict__ qlp,
                    const unsigned short* __restrict__ khp, const unsigned short* __restrict__ klp,
                    const unsigned short* __restrict__ vhp, const unsigned short* __restrict__ vlp,
                    unsigned short* __restrict__ aH, unsigned short* __restrict__ aL) {
  int lane = threadIdx.x & 31;
  int li = lane & 15;
  bool hiH = lane >= 16;
  int qb = blockIdx.x * 16;
  int h = blockIdx.y;

  // Q^T B-operand: lane = query i; d chunks {0..7,16..23} / {8..15,24..31}. Two 32-d sets.
  v16bf qh[2], ql[2];
  {
    size_t qoff = (size_t)(qb + li) * DMODEL + h * HDIM;
#pragma unroll
    for (int s = 0; s < 2; ++s) {
      int c = s * 32 + (hiH ? 8 : 0);
      qh[s] = ld8x2(qhp + qoff + c, qhp + qoff + c + 16);
      ql[s] = ld8x2(qlp + qoff + c, qlp + qoff + c + 16);
    }
  }

  v8f Oacc[4] = {};                 // O^T: d rows (4 chunks of 16), col = query i
  float mprev = -1e30f, lsum = 0.0f;
  const int i = qb + li;

  auto process_block = [&](int jb) {  // one 32-key block
    // ---- S^T = K_tile x Q^T : two 16-key subtiles, 3 term-accs each ----
    int jr0 = jb + li;      if (jr0 > SEQ - 1) jr0 = SEQ - 1;
    int jr1 = jb + 16 + li; if (jr1 > SEQ - 1) jr1 = SEQ - 1;
    size_t ko0 = (size_t)jr0 * DMODEL + h * HDIM;
    size_t ko1 = (size_t)jr1 * DMODEL + h * HDIM;
    v8f sa[6] = {};  // [sub*3 + term], round-robin distance 6
#pragma unroll
    for (int s = 0; s < 2; ++s) {
      int c = s * 32 + (hiH ? 8 : 0);
      v16bf kh0 = ld8x2(khp + ko0 + c, khp + ko0 + c + 16);
      v16bf kl0 = ld8x2(klp + ko0 + c, klp + ko0 + c + 16);
      v16bf kh1 = ld8x2(khp + ko1 + c, khp + ko1 + c + 16);
      v16bf kl1 = ld8x2(klp + ko1 + c, klp + ko1 + c + 16);
      sa[0] = WMMA_BF16(kh0, qh[s], sa[0]);
      sa[3] = WMMA_BF16(kh1, qh[s], sa[3]);
      sa[1] = WMMA_BF16(kh0, ql[s], sa[1]);
      sa[4] = WMMA_BF16(kh1, ql[s], sa[4]);
      sa[2] = WMMA_BF16(kl0, qh[s], sa[2]);
      sa[5] = WMMA_BF16(kl1, qh[s], sa[5]);
    }
    v8f st0 = sa[0] + sa[1] + sa[2];
    v8f st1 = sa[3] + sa[4] + sa[5];

    // ---- mask + online softmax (scale 1/sqrt(64) = 0.125) ----
    float s0[8], s1[8], mloc = -1e30f;
#pragma unroll
    for (int r = 0; r < 8; ++r) {
      int j0 = jb + (hiH ? 8 : 0) + r;
      int j1 = j0 + 16;
      float x0 = st0[r] * 0.125f;
      float x1 = st1[r] * 0.125f;
      if (!((j0 <= i) && ((j0 < NSINK) || (j0 + WINDOW > i)))) x0 = -1e30f;
      if (!((j1 <= i) && ((j1 < NSINK) || (j1 + WINDOW > i)))) x1 = -1e30f;
      s0[r] = x0; s1[r] = x1;
      mloc = fmaxf(mloc, fmaxf(x0, x1));
    }
    mloc = fmaxf(mloc, __shfl_xor(mloc, 16));
    float mnew = fmaxf(mprev, mloc);
    float alpha = __expf(mprev - mnew);
    float p0[8], p1[8], ssum = 0.0f;
#pragma unroll
    for (int r = 0; r < 8; ++r) {
      p0[r] = (s0[r] > -1e29f) ? __expf(s0[r] - mnew) : 0.0f;
      p1[r] = (s1[r] > -1e29f) ? __expf(s1[r] - mnew) : 0.0f;
      ssum += p0[r] + p1[r];
    }
    ssum += __shfl_xor(ssum, 16);
    lsum = lsum * alpha + ssum;
    mprev = mnew;
#pragma unroll
    for (int t = 0; t < 4; ++t)
#pragma unroll
      for (int r = 0; r < 8; ++r) Oacc[t][r] *= alpha;

    // ---- P^T as B operand (lane=i, K=j): half-swap shuffles + bf16 split ----
    v16bf ph, pl;
#pragma unroll
    for (int r = 0; r < 8; ++r) {
      float o0 = __shfl_xor(p0[r], 16);
      float o1 = __shfl_xor(p1[r], 16);
      float eA = hiH ? o1 : p0[r];   // element r     (K = base + r)
      float eB = hiH ? p1[r] : o0;   // element 8 + r (K = base + 8 + r)
      unsigned short ha = f2bf_rne(eA), hb = f2bf_rne(eB);
      ph[r]     = bfb(ha); pl[r]     = bfb(f2bf_rne(eA - bfbits2f(ha)));
      ph[8 + r] = bfb(hb); pl[8 + r] = bfb(f2bf_rne(eB - bfbits2f(hb)));
    }
    // ---- O^T += V^T x P^T : hoist V^T operands, term-major (distance 4) ----
    v16bf vh[4], vl[4];
#pragma unroll
    for (int d0 = 0; d0 < 4; ++d0) {
      size_t voff = (size_t)(h * HDIM + d0 * 16 + li) * SEQ + jb + (hiH ? 16 : 0);
      vh[d0] = ld16(vhp + voff);
      vl[d0] = ld16(vlp + voff);
    }
#pragma unroll
    for (int d0 = 0; d0 < 4; ++d0) Oacc[d0] = WMMA_BF16(vh[d0], ph, Oacc[d0]);
#pragma unroll
    for (int d0 = 0; d0 < 4; ++d0) Oacc[d0] = WMMA_BF16(vh[d0], pl, Oacc[d0]);
#pragma unroll
    for (int d0 = 0; d0 < 4; ++d0) Oacc[d0] = WMMA_BF16(vl[d0], ph, Oacc[d0]);
  };

  int wlo = qb - (WINDOW - 1);
  int jstart = (wlo > 0) ? (wlo & ~31) : 0;
  if (jstart > 0) process_block(0);                         // sink block (j<4)
  for (int jb = jstart; jb < qb + 16; jb += 32) process_block(jb);

  // ---- normalize + store bf16 hi/lo planes for the output projection ----
  float inv = 1.0f / lsum;
#pragma unroll
  for (int t = 0; t < 4; ++t) {
#pragma unroll
    for (int r = 0; r < 8; ++r) {
      int d = t * 16 + (hiH ? 8 : 0) + r;
      float v = Oacc[t][r] * inv;
      unsigned short hb = f2bf_rne(v);
      unsigned short lb = f2bf_rne(v - bfbits2f(hb));
      size_t o = (size_t)(qb + li) * DMODEL + h * HDIM + d;
      aH[o] = hb;
      aL[o] = lb;
    }
  }
}

// ============================================================================
// Launch. Workspace (u16 planes, 96 MB total -- L2 resident):
//  xh|xl (4M each) | 8 weight planes (1M each) | qh|ql|kh|kl (4M) | vTh|vTl (4M)
//  | attn h|l (4M)
// ============================================================================
extern "C" void kernel_launch(void* const* d_in, const int* in_sizes, int n_in,
                              void* d_out, int out_size, void* d_ws, size_t ws_size,
                              hipStream_t stream) {
  (void)in_sizes; (void)n_in; (void)out_size; (void)ws_size;
  const float* x  = (const float*)d_in[0];
  const float* Wq = (const float*)d_in[1];
  const float* Wk = (const float*)d_in[2];
  const float* Wv = (const float*)d_in[3];
  const float* Wo = (const float*)d_in[4];

  const size_t XP = (size_t)SEQ * DMODEL;     // 4M elements
  const size_t WP = (size_t)DMODEL * DMODEL;  // 1M elements
  unsigned short* ws = (unsigned short*)d_ws;
  unsigned short* xh  = ws;                  unsigned short* xl  = ws + XP;
  unsigned short* wqh = ws + 2 * XP;         unsigned short* wql = wqh + WP;
  unsigned short* wkh = wql + WP;            unsigned short* wkl = wkh + WP;
  unsigned short* wvh = wkl + WP;            unsigned short* wvl = wvh + WP;
  unsigned short* woh = wvl + WP;            unsigned short* wol = woh + WP;
  unsigned short* qh  = wol + WP;            unsigned short* ql  = qh + XP;
  unsigned short* kh  = ql + XP;             unsigned short* kl  = kh + XP;
  unsigned short* vTh = kl + XP;             unsigned short* vTl = vTh + XP;
  unsigned short* ath = vTl + XP;            unsigned short* atl = ath + XP;

  // one-time splits
  split_f32_bf16x2<<<(int)(XP / 8 / 256), 256, 0, stream>>>(x,  xh,  xl,  (int)(XP / 8));
  split_f32_bf16x2<<<(int)(WP / 8 / 256), 256, 0, stream>>>(Wq, wqh, wql, (int)(WP / 8));
  split_f32_bf16x2<<<(int)(WP / 8 / 256), 256, 0, stream>>>(Wk, wkh, wkl, (int)(WP / 8));
  split_f32_bf16x2<<<(int)(WP / 8 / 256), 256, 0, stream>>>(Wv, wvh, wvl, (int)(WP / 8));
  split_f32_bf16x2<<<(int)(WP / 8 / 256), 256, 0, stream>>>(Wo, woh, wol, (int)(WP / 8));

  // 2048 wave-tiles per GEMM, 8 waves per block -> 256 blocks
  gemm_nt_bf16x3<<<256, 256, 0, stream>>>(xh, xl, wqh, wql, nullptr, qh,  ql,  1);
  gemm_nt_bf16x3<<<256, 256, 0, stream>>>(xh, xl, wkh, wkl, nullptr, kh,  kl,  1);
  gemm_nt_bf16x3<<<256, 256, 0, stream>>>(xh, xl, wvh, wvl, nullptr, vTh, vTl, 2);

  attn_streaming_wmma<<<dim3(SEQ / 16, NHEADS), 32, 0, stream>>>(qh, ql, kh, kl, vTh, vTl, ath, atl);

  gemm_nt_bf16x3<<<256, 256, 0, stream>>>(ath, atl, woh, wol, (float*)d_out, nullptr, nullptr, 0);
}
